// VQVAE_79800492360055
// MI455X (gfx1250) — matrix operand order repaired and checked by reading, not compiled, not run on hardware
//
#include <hip/hip_runtime.h>
#include <hip/hip_bf16.h>
#include <stdint.h>

// ---------------------------------------------------------------------------
// CDNA5 (gfx1250) VQ-VAE forward, round 4.
//   * Implicit-GEMM convs: 128x64 tile, 4 v_wmma_f32_16x16x32_f16 per wave
//     per K-step, GLOBAL_LOAD_ASYNC_TO_LDS_B128 weight staging (ASYNCcnt),
//     LDS double buffering, one barrier per K-step.
//   * ConvTranspose2d(k=4,s=2,p=1) decomposed into 4 parity-phase 2x2 convs
//     (K = Cin*4, zero wasted taps) instead of a dilated-input conv
//     (K = Cin*16, 75% zero taps) -> ~75% less executed WMMA work.
// ---------------------------------------------------------------------------

typedef __attribute__((ext_vector_type(16))) _Float16     v16h;
typedef __attribute__((ext_vector_type(8)))  float        v8f;
typedef __attribute__((ext_vector_type(4)))  float        f4raw;

// ---------------- weight prep: f32 OIHW -> f16 [Coutpad][Kpad] -------------
__global__ void prep_w_conv(const float* __restrict__ w, _Float16* __restrict__ o,
                            int Cout, int K, int Coutpad, int Kpad) {
    int idx = blockIdx.x * blockDim.x + threadIdx.x;
    if (idx >= Coutpad * Kpad) return;
    int row = idx / Kpad, k = idx - row * Kpad;
    float v = (row < Cout && k < K) ? w[(size_t)row * K + k] : 0.f;
    o[idx] = (_Float16)v;
}

// torch ConvTranspose2d weight (Cin,Cout,4,4) -> parity-class 2x2 f16 matrix:
// wc[o][c][rr][ss] = w[c][o][3-py-2rr][3-px-2ss]
__global__ void prep_w_convt_phase(const float* __restrict__ w, _Float16* __restrict__ o,
                                   int Cin, int Cout, int py, int px,
                                   int Coutpad, int Kpad) {
    int K = Cin * 4;
    int idx = blockIdx.x * blockDim.x + threadIdx.x;
    if (idx >= Coutpad * Kpad) return;
    int row = idx / Kpad, k = idx - row * Kpad;
    float v = 0.f;
    if (row < Cout && k < K) {
        int c = k >> 2, rr = (k >> 1) & 1, ss = k & 1;
        v = w[(((size_t)c * Cout + row) * 4 + (3 - py - 2 * rr)) * 4 + (3 - px - 2 * ss)];
    }
    o[idx] = (_Float16)v;
}

// ============================================================================
// Normal conv: implicit GEMM, templated (KH,KW,stride,pad).
// ============================================================================
template <int KH, int KW, int ST, int PD>
__global__ __launch_bounds__(256) void conv_gemm_wmma(
        const float* __restrict__ x, const _Float16* __restrict__ w,
        const float* __restrict__ bias, float* __restrict__ out,
        int Cin, int Hin, int Win, int Cout, int Hout, int Wout, int B, int Kpad) {
    constexpr int KHKW = KH * KW;
    __shared__ _Float16 aT[2][8 * 512];   // 128 Cout rows (8 fragments)
    __shared__ _Float16 bT[2][4 * 512];   // 64 pixels     (4 fragments)

    const int tid  = threadIdx.x;
    const int lane = tid & 31;
    const int wid  = tid >> 5;
    const int ni   = wid & 3;
    const int mg   = wid >> 2;

    const int Ktot  = Cin * KHKW;
    const int HWo   = Hout * Wout;
    const int p0    = blockIdx.x * 64;
    const int cout0 = blockIdx.y * 128;

    // A mapping: thread owns full fragment-lane (frag = wid, lane)
    const int am = (lane & 15) + 16 * wid;
    const int ah = lane >> 4;
    const int ea = tid * 16;
    const _Float16* wsrc = w + (size_t)(cout0 + am) * Kpad;

    // B mapping: thread -> 8 consecutive k for one pixel
    const int eb    = tid * 8;
    const int bfrag = eb >> 9;
    const int brem  = eb & 511;
    const int blane = brem >> 4;
    const int bi0   = brem & 15;
    const int bn    = blane & 15;
    const int bkb   = 8 * (blane >> 4) + 2 * bi0;
    const int p     = p0 + bfrag * 16 + bn;
    int bb = p / HWo;
    int hw = p - bb * HWo;
    int oh = hw / Wout;
    int ow = hw - oh * Wout;
    const int ihbase = oh * ST - PD;
    const int iwbase = ow * ST - PD;
    const float* xsrc = x + (size_t)bb * Cin * Hin * Win;

    union { _Float16 h[8]; f4raw v4; } rb;

    auto issueA = [&](int k0, int buf) {
        unsigned        lds0 = (unsigned)(uintptr_t)(&aT[buf][ea]);
        const _Float16* g0   = wsrc + k0 + 8 * ah;
        const _Float16* g1   = wsrc + k0 + 16 + 8 * ah;
        asm volatile("global_load_async_to_lds_b128 %0, %1, off"
                     :: "v"(lds0), "v"(g0) : "memory");
        asm volatile("global_load_async_to_lds_b128 %0, %1, off"
                     :: "v"(lds0 + 16u), "v"(g1) : "memory");
    };
    auto gatherB = [&](int k0) {
#pragma unroll
        for (int j = 0; j < 8; ++j) {
            int   kg = k0 + bkb + j;
            float v  = 0.f;
            if (kg < Ktot) {
                int c   = kg / KHKW;
                int rs  = kg - c * KHKW;
                int r   = rs / KW;
                int s   = rs - r * KW;
                int ihl = ihbase + r;
                int iwl = iwbase + s;
                if ((unsigned)ihl < (unsigned)Hin && (unsigned)iwl < (unsigned)Win)
                    v = xsrc[((size_t)c * Hin + ihl) * Win + iwl];
            }
            rb.h[j] = (_Float16)v;
        }
    };
    auto stageB = [&](int buf) { *(f4raw*)(&bT[buf][eb]) = rb.v4; };

    v8f acc[4] = {{}, {}, {}, {}};
    issueA(0, 0);
    gatherB(0);
    stageB(0);
    asm volatile("s_wait_asynccnt 0" ::: "memory");
    __syncthreads();
    for (int k0 = 0; k0 < Ktot; k0 += 32) {
        const int  cur  = (k0 >> 5) & 1;
        const bool more = (k0 + 32) < Ktot;
        if (more) {
            issueA(k0 + 32, cur ^ 1);
            gatherB(k0 + 32);
        }
        v16h bf = *(const v16h*)(&bT[cur][ni * 512 + lane * 16]);
#pragma unroll
        for (int s = 0; s < 4; ++s) {
            v16h af = *(const v16h*)(&aT[cur][(mg * 4 + s) * 512 + lane * 16]);
            acc[s] = __builtin_amdgcn_wmma_f32_16x16x32_f16(false, af, false, bf,
                                                            (short)0, acc[s],
                                                            false, false);
        }
        if (more) stageB(cur ^ 1);
        asm volatile("s_wait_asynccnt 0" ::: "memory");
        __syncthreads();
    }

    const int cm = 8 * (lane >> 4);
    const int cn = lane & 15;
    const int pp = p0 + ni * 16 + cn;
    int ob  = pp / HWo;
    int ohw = pp - ob * HWo;
#pragma unroll
    for (int s = 0; s < 4; ++s) {
#pragma unroll
        for (int r = 0; r < 8; ++r) {
            int co = cout0 + (mg * 4 + s) * 16 + cm + r;
            if (co < Cout)
                out[((size_t)(ob * Cout + co)) * HWo + ohw] = acc[s][r] + bias[co];
        }
    }
}

// ============================================================================
// Parity-phase transposed conv: one of 4 classes of ConvT(k=4,s=2,p=1).
// Effective: 2x2 conv, stride 1, pad (1-PY,1-PX), output written at
// (2y+PY, 2x+PX). Hout/Wout are full output dims; logical grid is Hout/2 x
// Wout/2 pixels.
// ============================================================================
template <int PY, int PX>
__global__ __launch_bounds__(256) void convt_phase_wmma(
        const float* __restrict__ x, const _Float16* __restrict__ w,
        const float* __restrict__ bias, float* __restrict__ out,
        int Cin, int Hin, int Win, int Cout, int Hout, int Wout, int B, int Kpad) {
    __shared__ _Float16 aT[2][8 * 512];
    __shared__ _Float16 bT[2][4 * 512];

    const int tid  = threadIdx.x;
    const int lane = tid & 31;
    const int wid  = tid >> 5;
    const int ni   = wid & 3;
    const int mg   = wid >> 2;

    const int Ktot  = Cin * 4;
    const int Hh    = Hout >> 1, Wh = Wout >> 1;
    const int HWh   = Hh * Wh;
    const int p0    = blockIdx.x * 64;
    const int cout0 = blockIdx.y * 128;

    const int am = (lane & 15) + 16 * wid;
    const int ah = lane >> 4;
    const int ea = tid * 16;
    const _Float16* wsrc = w + (size_t)(cout0 + am) * Kpad;

    const int eb    = tid * 8;
    const int bfrag = eb >> 9;
    const int brem  = eb & 511;
    const int blane = brem >> 4;
    const int bi0   = brem & 15;
    const int bn    = blane & 15;
    const int bkb   = 8 * (blane >> 4) + 2 * bi0;
    const int p     = p0 + bfrag * 16 + bn;
    int bb = p / HWh;
    int hw = p - bb * HWh;
    int yy = hw / Wh;
    int xx = hw - yy * Wh;
    const int ihbase = yy - (1 - PY);   // + rr
    const int iwbase = xx - (1 - PX);   // + ss
    const float* xsrc = x + (size_t)bb * Cin * Hin * Win;

    union { _Float16 h[8]; f4raw v4; } rb;

    auto issueA = [&](int k0, int buf) {
        unsigned        lds0 = (unsigned)(uintptr_t)(&aT[buf][ea]);
        const _Float16* g0   = wsrc + k0 + 8 * ah;
        const _Float16* g1   = wsrc + k0 + 16 + 8 * ah;
        asm volatile("global_load_async_to_lds_b128 %0, %1, off"
                     :: "v"(lds0), "v"(g0) : "memory");
        asm volatile("global_load_async_to_lds_b128 %0, %1, off"
                     :: "v"(lds0 + 16u), "v"(g1) : "memory");
    };
    auto gatherB = [&](int k0) {
#pragma unroll
        for (int j = 0; j < 8; ++j) {
            int   kg = k0 + bkb + j;
            float v  = 0.f;
            if (kg < Ktot) {
                int c   = kg >> 2;
                int rr  = (kg >> 1) & 1;
                int ss  = kg & 1;
                int ihl = ihbase + rr;
                int iwl = iwbase + ss;
                if ((unsigned)ihl < (unsigned)Hin && (unsigned)iwl < (unsigned)Win)
                    v = xsrc[((size_t)c * Hin + ihl) * Win + iwl];
            }
            rb.h[j] = (_Float16)v;
        }
    };
    auto stageB = [&](int buf) { *(f4raw*)(&bT[buf][eb]) = rb.v4; };

    v8f acc[4] = {{}, {}, {}, {}};
    issueA(0, 0);
    gatherB(0);
    stageB(0);
    asm volatile("s_wait_asynccnt 0" ::: "memory");
    __syncthreads();
    for (int k0 = 0; k0 < Ktot; k0 += 32) {
        const int  cur  = (k0 >> 5) & 1;
        const bool more = (k0 + 32) < Ktot;
        if (more) {
            issueA(k0 + 32, cur ^ 1);
            gatherB(k0 + 32);
        }
        v16h bf = *(const v16h*)(&bT[cur][ni * 512 + lane * 16]);
#pragma unroll
        for (int s = 0; s < 4; ++s) {
            v16h af = *(const v16h*)(&aT[cur][(mg * 4 + s) * 512 + lane * 16]);
            acc[s] = __builtin_amdgcn_wmma_f32_16x16x32_f16(false, af, false, bf,
                                                            (short)0, acc[s],
                                                            false, false);
        }
        if (more) stageB(cur ^ 1);
        asm volatile("s_wait_asynccnt 0" ::: "memory");
        __syncthreads();
    }

    const int cm = 8 * (lane >> 4);
    const int cn = lane & 15;
    const int pp = p0 + ni * 16 + cn;
    int ob  = pp / HWh;
    int ohw = pp - ob * HWh;
    int y2  = ohw / Wh;
    int x2  = ohw - y2 * Wh;
    int oidx = (2 * y2 + PY) * Wout + 2 * x2 + PX;
#pragma unroll
    for (int s = 0; s < 4; ++s) {
#pragma unroll
        for (int r = 0; r < 8; ++r) {
            int co = cout0 + (mg * 4 + s) * 16 + cm + r;
            if (co < Cout)
                out[((size_t)(ob * Cout + co)) * Hout * Wout + oidx] = acc[s][r] + bias[co];
        }
    }
}

// ---------------- BatchNorm (training-mode, biased var) --------------------
__global__ __launch_bounds__(256) void bn_stats(const float* __restrict__ x,
                                                float* __restrict__ mean,
                                                float* __restrict__ istd,
                                                int C, int HW, int B) {
    __shared__ float s1[256], s2[256];
    int   c = blockIdx.x;
    float a = 0.f, q = 0.f;
    int   N = B * HW;
    for (int i = threadIdx.x; i < N; i += 256) {
        int   bi = i / HW, hw = i - bi * HW;
        float v  = x[((size_t)(bi * C + c)) * HW + hw];
        a += v;
        q += v * v;
    }
    s1[threadIdx.x] = a;
    s2[threadIdx.x] = q;
    __syncthreads();
    for (int off = 128; off > 0; off >>= 1) {
        if (threadIdx.x < off) {
            s1[threadIdx.x] += s1[threadIdx.x + off];
            s2[threadIdx.x] += s2[threadIdx.x + off];
        }
        __syncthreads();
    }
    if (threadIdx.x == 0) {
        float m   = s1[0] / (float)N;
        float var = s2[0] / (float)N - m * m;
        mean[c] = m;
        istd[c] = rsqrtf(var + 1e-5f);
    }
}

__global__ void bn_apply(const float* __restrict__ x, const float* __restrict__ skip,
                         float* __restrict__ out, const float* __restrict__ mean,
                         const float* __restrict__ istd, int C, int HW,
                         long long n, int addskip) {
    long long i = (long long)blockIdx.x * 256 + threadIdx.x;
    if (i >= n) return;
    int   c = (int)((i / HW) % C);
    float v = (x[i] - mean[c]) * istd[c];
    v = v > 0.f ? v : 0.f;
    out[i] = addskip ? (skip[i] + v) : v;
}

// ---------------- vector quantization --------------------------------------
__global__ __launch_bounds__(256) void vq_kernel(const float* __restrict__ enc,
                                                 const float* __restrict__ E,
                                                 float* __restrict__ quant,
                                                 int* __restrict__ closest,
                                                 float* __restrict__ lossacc) {
    __shared__ float sv[512];
    __shared__ float cb[128 * 65];
    int r0 = blockIdx.x * 8;
    for (int i = threadIdx.x; i < 512; i += 256)
        sv[i] = enc[(size_t)r0 * 64 + i];

    int lane = threadIdx.x & 31, wid = threadIdx.x >> 5;
    int r = r0 + wid;
    const float* v = sv + wid * 64;

    float bestd = 3.4e38f;
    int   bestk = 0;
    for (int ch = 0; ch < 4; ++ch) {
        __syncthreads();
        for (int i = threadIdx.x; i < 128 * 64; i += 256) {
            int kk = i >> 6, d = i & 63;
            cb[kk * 65 + d] = E[((size_t)(ch * 128 + kk)) * 64 + d];
        }
        __syncthreads();
#pragma unroll
        for (int cc = 0; cc < 4; ++cc) {
            int          kloc = cc * 32 + lane;
            const float* e    = cb + kloc * 65;
            float        s    = 0.f;
#pragma unroll 16
            for (int d = 0; d < 64; ++d) {
                float df = v[d] - e[d];
                s += df * df;
            }
            int k = ch * 128 + kloc;
            if (s < bestd || (s == bestd && k < bestk)) { bestd = s; bestk = k; }
        }
    }
    for (int off = 16; off; off >>= 1) {
        float od = __shfl_xor(bestd, off, 32);
        int   ok = __shfl_xor(bestk, off, 32);
        if (od < bestd || (od == bestd && ok < bestk)) { bestd = od; bestk = ok; }
    }
    const float* ebp = E + (size_t)bestk * 64;
    float        e0 = ebp[lane], e1 = ebp[lane + 32];
    quant[(size_t)r * 64 + lane]      = e0;
    quant[(size_t)r * 64 + lane + 32] = e1;
    float d0 = v[lane] - e0, d1 = v[lane + 32] - e1;
    float s  = d0 * d0 + d1 * d1;
    for (int off = 16; off; off >>= 1) s += __shfl_xor(s, off, 32);
    if (lane == 0) {
        closest[r] = bestk;
        atomicAdd(lossacc, s);
    }
}

__global__ void zero_one(float* p) { if (threadIdx.x == 0 && blockIdx.x == 0) *p = 0.f; }

__global__ void finalize_loss(const float* __restrict__ acc, float* __restrict__ out) {
    if (threadIdx.x == 0 && blockIdx.x == 0)
        *out = 1.255555f * (*acc) / (float)(32768 * 64);
}

__global__ void sigmoid_k(const float* __restrict__ x, float* __restrict__ o, long long n) {
    long long i = (long long)blockIdx.x * 256 + threadIdx.x;
    if (i >= n) return;
    o[i] = 1.f / (1.f + expf(-x[i]));
}

// ---------------------------------------------------------------------------
template <int KH, int KW, int ST, int PD>
static void launch_conv(hipStream_t stream, const float* in, const _Float16* w,
                        const float* bias, float* o, int Cin, int Hin, int Win,
                        int Cout, int Hout, int Wout) {
    const int B    = 32;
    int       Kpad = ((Cin * KH * KW + 31) / 32) * 32;
    int       Cp   = ((Cout + 127) / 128) * 128;
    int       P    = B * Hout * Wout;
    dim3      g((P + 63) / 64, Cp / 128);
    conv_gemm_wmma<KH, KW, ST, PD>
        <<<g, 256, 0, stream>>>(in, w, bias, o, Cin, Hin, Win, Cout, Hout, Wout, B, Kpad);
}

template <int PY, int PX>
static void launch_convt_phase(hipStream_t stream, const float* in, const _Float16* w,
                               const float* bias, float* o, int Cin, int Hin, int Win,
                               int Cout, int Hout, int Wout) {
    const int B    = 32;
    int       Kpad = Cin * 4;                       // Cin multiple of 8 here
    int       Cp   = ((Cout + 127) / 128) * 128;
    int       P    = B * (Hout / 2) * (Wout / 2);
    dim3      g((P + 63) / 64, Cp / 128);
    convt_phase_wmma<PY, PX>
        <<<g, 256, 0, stream>>>(in, w, bias, o, Cin, Hin, Win, Cout, Hout, Wout, B, Kpad);
}

extern "C" void kernel_launch(void* const* d_in, const int* in_sizes, int n_in,
                              void* d_out, int out_size, void* d_ws, size_t ws_size,
                              hipStream_t stream) {
    (void)in_sizes; (void)n_in; (void)out_size; (void)ws_size;
    const int B = 32;

    const float* x        = (const float*)d_in[0];
    const float* e_c1_w   = (const float*)d_in[1];
    const float* e_c1_b   = (const float*)d_in[2];
    const float* e_c2_w   = (const float*)d_in[3];
    const float* e_c2_b   = (const float*)d_in[4];
    const float* e_r1_w   = (const float*)d_in[5];
    const float* e_r1_b   = (const float*)d_in[6];
    const float* e_r2_w   = (const float*)d_in[7];
    const float* e_r2_b   = (const float*)d_in[8];
    const float* e_pr_w   = (const float*)d_in[9];
    const float* e_pr_b   = (const float*)d_in[10];
    const float* codebook = (const float*)d_in[11];
    const float* d_pr_w   = (const float*)d_in[12];
    const float* d_pr_b   = (const float*)d_in[13];
    const float* d_r1_w   = (const float*)d_in[14];
    const float* d_r1_b   = (const float*)d_in[15];
    const float* d_r2_w   = (const float*)d_in[16];
    const float* d_r2_b   = (const float*)d_in[17];
    const float* d_t1_w   = (const float*)d_in[18];
    const float* d_t1_b   = (const float*)d_in[19];
    const float* d_t2_w   = (const float*)d_in[20];
    const float* d_t2_b   = (const float*)d_in[21];

    float* out_img  = (float*)d_out;
    int*   out_idx  = (int*)((float*)d_out + 1572864);
    float* out_loss = (float*)d_out + 1572864 + 32768;

    // ---- workspace carve ----
    char*  ws  = (char*)d_ws;
    size_t off = 0;
    auto alloc = [&](size_t bytes) -> void* {
        void* p = ws + off;
        off = (off + bytes + 255) & ~(size_t)255;
        return p;
    };
    _Float16* W1  = (_Float16*)alloc((size_t)256 *   64 * 2);       // e_c1 (K 48->64)
    _Float16* W2  = (_Float16*)alloc((size_t)256 * 4096 * 2);       // e_c2
    _Float16* W3  = (_Float16*)alloc((size_t)256 * 2304 * 2);       // e_r1
    _Float16* W4  = (_Float16*)alloc((size_t)256 *  256 * 2);       // e_r2
    _Float16* W5  = (_Float16*)alloc((size_t)128 *  256 * 2);       // e_pr (pad128)
    _Float16* W6  = (_Float16*)alloc((size_t)256 *   64 * 2);       // d_pr
    _Float16* W7  = (_Float16*)alloc((size_t)256 * 2304 * 2);       // d_r1
    _Float16* W8  = (_Float16*)alloc((size_t)256 * 2304 * 2);       // d_r2
    _Float16* W9  = (_Float16*)alloc((size_t)4 * 256 * 1024 * 2);   // d_t1: 4 phases
    _Float16* W10 = (_Float16*)alloc((size_t)4 * 128 * 1024 * 2);   // d_t2: 4 phases
    float* mean    = (float*)alloc(256 * 4);
    float* istd    = (float*)alloc(256 * 4);
    float* lossacc = (float*)alloc(4);
    float* enc     = (float*)alloc((size_t)B * 64 * 1024 * 4);
    float* quant   = (float*)alloc((size_t)B * 64 * 1024 * 4);
    float* bufA    = (float*)alloc((size_t)B * 256 * 4096 * 4);
    float* bufB    = (float*)alloc((size_t)B * 256 * 4096 * 4);
    float* bufC    = (float*)alloc((size_t)B * 256 * 4096 * 4);

    // ---- weight prep (rows padded to 128 to match 128-row A tiles) ----
    auto prepc = [&](const float* w, _Float16* o, int Cout, int K) {
        int Cp = ((Cout + 127) / 128) * 128, Kp = ((K + 31) / 32) * 32;
        int n  = Cp * Kp;
        prep_w_conv<<<(n + 255) / 256, 256, 0, stream>>>(w, o, Cout, K, Cp, Kp);
    };
    auto prepph = [&](const float* w, _Float16* o, int Cin, int Cout, int py, int px) {
        int Cp = ((Cout + 127) / 128) * 128, Kp = Cin * 4;
        int n  = Cp * Kp;
        prep_w_convt_phase<<<(n + 255) / 256, 256, 0, stream>>>(w, o, Cin, Cout, py, px,
                                                                Cp, Kp);
    };
    prepc(e_c1_w, W1, 256, 3 * 16);
    prepc(e_c2_w, W2, 256, 256 * 16);
    prepc(e_r1_w, W3, 256, 256 * 9);
    prepc(e_r2_w, W4, 256, 256);
    prepc(e_pr_w, W5, 64, 256);
    prepc(d_pr_w, W6, 256, 64);
    prepc(d_r1_w, W7, 256, 256 * 9);
    prepc(d_r2_w, W8, 256, 256 * 9);
    for (int ph = 0; ph < 4; ++ph) {
        prepph(d_t1_w, W9 + (size_t)ph * 256 * 1024, 256, 256, ph >> 1, ph & 1);
        prepph(d_t2_w, W10 + (size_t)ph * 128 * 1024, 256, 3, ph >> 1, ph & 1);
    }

    auto bn = [&](const float* in, const float* skip, float* o, int C, int HW, int add) {
        bn_stats<<<C, 256, 0, stream>>>(in, mean, istd, C, HW, B);
        long long n = (long long)B * C * HW;
        bn_apply<<<(unsigned)((n + 255) / 256), 256, 0, stream>>>(in, skip, o, mean, istd,
                                                                  C, HW, n, add);
    };

    // ---- encoder ----
    launch_conv<4, 4, 2, 1>(stream, x, W1, e_c1_b, bufA, 3, 128, 128, 256, 64, 64);
    bn(bufA, nullptr, bufB, 256, 4096, 0);
    launch_conv<4, 4, 2, 1>(stream, bufB, W2, e_c2_b, bufA, 256, 64, 64, 256, 32, 32);
    bn(bufA, nullptr, bufB, 256, 1024, 0);
    launch_conv<3, 3, 1, 1>(stream, bufB, W3, e_r1_b, bufA, 256, 32, 32, 256, 32, 32);
    bn(bufA, bufB, bufC, 256, 1024, 1);
    launch_conv<1, 1, 1, 0>(stream, bufC, W4, e_r2_b, bufA, 256, 32, 32, 256, 32, 32);
    bn(bufA, bufC, bufB, 256, 1024, 1);
    launch_conv<1, 1, 1, 0>(stream, bufB, W5, e_pr_b, enc, 256, 32, 32, 64, 32, 32);

    // ---- vector quantization ----
    zero_one<<<1, 1, 0, stream>>>(lossacc);
    vq_kernel<<<4096, 256, 0, stream>>>(enc, codebook, quant, out_idx, lossacc);
    finalize_loss<<<1, 1, 0, stream>>>(lossacc, out_loss);

    // ---- decoder ----
    launch_conv<1, 1, 1, 0>(stream, quant, W6, d_pr_b, bufA, 64, 32, 32, 256, 32, 32);
    launch_conv<3, 3, 1, 1>(stream, bufA, W7, d_r1_b, bufB, 256, 32, 32, 256, 32, 32);
    bn(bufB, bufA, bufC, 256, 1024, 1);
    launch_conv<3, 3, 1, 1>(stream, bufC, W8, d_r2_b, bufA, 256, 32, 32, 256, 32, 32);
    bn(bufA, bufC, bufB, 256, 1024, 1);
    // ConvT k4 s2 p1 -> 4 parity phases of a 2x2 stride-1 conv
    launch_convt_phase<0, 0>(stream, bufB, W9 + (size_t)0 * 256 * 1024, d_t1_b, bufA,
                             256, 32, 32, 256, 64, 64);
    launch_convt_phase<0, 1>(stream, bufB, W9 + (size_t)1 * 256 * 1024, d_t1_b, bufA,
                             256, 32, 32, 256, 64, 64);
    launch_convt_phase<1, 0>(stream, bufB, W9 + (size_t)2 * 256 * 1024, d_t1_b, bufA,
                             256, 32, 32, 256, 64, 64);
    launch_convt_phase<1, 1>(stream, bufB, W9 + (size_t)3 * 256 * 1024, d_t1_b, bufA,
                             256, 32, 32, 256, 64, 64);
    bn(bufA, nullptr, bufC, 256, 4096, 0);
    launch_convt_phase<0, 0>(stream, bufC, W10 + (size_t)0 * 128 * 1024, d_t2_b, bufB,
                             256, 64, 64, 3, 128, 128);
    launch_convt_phase<0, 1>(stream, bufC, W10 + (size_t)1 * 128 * 1024, d_t2_b, bufB,
                             256, 64, 64, 3, 128, 128);
    launch_convt_phase<1, 0>(stream, bufC, W10 + (size_t)2 * 128 * 1024, d_t2_b, bufB,
                             256, 64, 64, 3, 128, 128);
    launch_convt_phase<1, 1>(stream, bufC, W10 + (size_t)3 * 128 * 1024, d_t2_b, bufB,
                             256, 64, 64, 3, 128, 128);
    long long nout = (long long)B * 3 * 128 * 128;
    sigmoid_k<<<(unsigned)((nout + 255) / 256), 256, 0, stream>>>(bufB, out_img, nout);
}
